// Model_19121194402216
// MI455X (gfx1250) — matrix operand order repaired and checked by reading, not compiled, not run on hardware
//
#include <hip/hip_runtime.h>
#include <hip/hip_bf16.h>

// ---------------------------------------------------------------------------
// STP-RNN (Masse et al.) fused scan for gfx1250.
//   T=200, B=512, N_IN=128, H=512, N_OUT=32.
// 32 persistent WGs x 16 batch rows (one WMMA M-tile); h/sx/su stay on-chip.
// GEMMs via v_wmma_f32_16x16x32_f16; weights pre-transposed/ReLU'd/EI-scaled
// to f16 (L2-resident, streamed per step). Output GEMM fused per step so
// hs[T,B,H] never round-trips HBM. Weight loads carry a runtime-zero offset
// (zoff*t) to stop LICM from hoisting 200 iterations of B-fragments into
// registers (which previously spilled to scratch).
// ---------------------------------------------------------------------------

#define T_STEPS 200
#define BATCH   512
#define NIN     128
#define HDIM    512
#define NOUT    32
#define BT      16              // batch rows per workgroup (one M-tile)
#define NWG     (BATCH / BT)    // 32 workgroups
#define THREADS 256             // 8 waves of 32
#define EPT     (BT * HDIM / THREADS)  // 32 state elements per thread

typedef __attribute__((ext_vector_type(16))) _Float16 v16h;
typedef __attribute__((ext_vector_type(8)))  _Float16 v8h;
typedef __attribute__((ext_vector_type(8)))  float    v8f;

// ---- weight prep: fold ReLU (+EI sign on K for w_rnn), transpose to [N][K] f16
__global__ void prep_weights(const float* __restrict__ w_in,   // [NIN,H]
                             const float* __restrict__ w_rnn,  // [H,H]
                             const float* __restrict__ w_out,  // [H,NOUT]
                             _Float16* __restrict__ winT,      // [H][NIN]
                             _Float16* __restrict__ wrnnT,     // [H][H]
                             _Float16* __restrict__ woutT)     // [NOUT][H]
{
    int idx = blockIdx.x * blockDim.x + threadIdx.x;
    if (idx < HDIM * NIN) {
        int n = idx / NIN, k = idx % NIN;
        float v = w_in[k * HDIM + n];
        winT[idx] = (_Float16)(v > 0.f ? v : 0.f);
    } else if (idx < HDIM * NIN + HDIM * HDIM) {
        int r = idx - HDIM * NIN;
        int n = r / HDIM, k = r % HDIM;
        // w_rnn_eff[i][j] = EI[i] * relu(w_rnn[i][j]); EI[i] = +1 for i<409 else -1
        float ei = (k < 409) ? 1.f : -1.f;
        float v = w_rnn[k * HDIM + n];
        wrnnT[r] = (_Float16)(ei * (v > 0.f ? v : 0.f));
    } else if (idx < HDIM * NIN + HDIM * HDIM + NOUT * HDIM) {
        int r = idx - HDIM * NIN - HDIM * HDIM;
        int n = r / HDIM, k = r % HDIM;
        float v = w_out[k * NOUT + n];
        woutT[r] = (_Float16)(v > 0.f ? v : 0.f);
    }
}

// A fragment (16-bit, 16x32): row-major [row][K]; per lane K in
// {8h..8h+7} U {16+8h..23+8h}  (ISA 7.12.2 table)
__device__ __forceinline__ v16h load_a_frag(const _Float16* rowbase, int kt, int half) {
    const _Float16* p = rowbase + kt * 32 + 8 * half;
    v8h lo = *(const v8h*)p;
    v8h hi = *(const v8h*)(p + 16);
    return __builtin_shufflevector(lo, hi, 0,1,2,3,4,5,6,7,8,9,10,11,12,13,14,15);
}

// B fragment (16-bit, 32x16) from transposed weights WT[n][K]; per lane K in
// {8h..8h+7} U {16+8h..23+8h}  (same as A fragment)
__device__ __forceinline__ v16h load_b_frag(const _Float16* rowbase, int kt, int half) {
    const _Float16* p = rowbase + kt * 32 + 8 * half;
    v8h lo = *(const v8h*)p;
    v8h hi = *(const v8h*)(p + 16);
    return __builtin_shufflevector(lo, hi, 0,1,2,3,4,5,6,7,8,9,10,11,12,13,14,15);
}

__global__ __launch_bounds__(THREADS)
void stp_rnn(const float* __restrict__ g_x,      // [T,B,NIN]
             const float* __restrict__ g_noise,  // [T,B,H]
             const _Float16* __restrict__ winT,  // [H][NIN]
             const _Float16* __restrict__ wrnnT, // [H][H]
             const _Float16* __restrict__ woutT, // [NOUT][H]
             const float* __restrict__ b_rnn,    // [H]
             const float* __restrict__ b_out,    // [NOUT]
             const float* __restrict__ h0,       // [H]
             const float* __restrict__ sx0,      // [H]
             const float* __restrict__ su0,      // [H]
             float* __restrict__ g_y,            // [T,B,NOUT]
             int zoff)                           // always 0; defeats LICM hoist
{
    __shared__ float    s_h [BT][HDIM];   // 32 KB  hidden state (f32)
    __shared__ _Float16 s_hp[BT][HDIM];   // 16 KB  h_post (f16); reused as h_f16 for Y
    __shared__ _Float16 s_x [BT][NIN];    //  4 KB  x_t (f16)

    const int tid  = threadIdx.x;
    const int lane = tid & 31;
    const int wave = tid >> 5;
    const int half = lane >> 4;
    const int l16  = lane & 15;
    const int b0   = blockIdx.x * BT;

    // --- per-thread register state for sx/su (fixed partition i = c*256+tid) ---
    float rsx[EPT], rsu[EPT];
    #pragma unroll
    for (int c = 0; c < EPT; ++c) {
        int i = c * THREADS + tid;
        int n = i & (HDIM - 1);
        rsx[c] = sx0[n];
        rsu[c] = su0[n];
        ((float*)s_h)[i] = h0[n];
    }
    __syncthreads();

    // wave w owns hidden columns [w*64, w*64+64): 4 N-tiles
    const int nbase = wave * 64;
    float bias[4];
    #pragma unroll
    for (int j = 0; j < 4; ++j) bias[j] = b_rnn[nbase + j * 16 + l16];
    const float bouty = (wave < 2) ? b_out[wave * 16 + l16] : 0.f;

    for (int t = 0; t < T_STEPS; ++t) {
        // t-dependent (but actually zero) offset: keeps weight loads in-loop
        const size_t zo = (size_t)(zoff * t);
        const _Float16* wiT = winT  + zo;
        const _Float16* wrT = wrnnT + zo;
        const _Float16* woT = woutT + zo;

        // ================= Phase E: STP elementwise + staging =================
        // prefetch this step's noise slice (one contiguous 32KB block per WG)
        __builtin_prefetch(g_noise + ((size_t)t * BATCH + b0) * HDIM + tid * 32, 0, 0);
        if (tid < 64)  // x_t slice: contiguous 8KB
            __builtin_prefetch(g_x + ((size_t)t * BATCH + b0) * NIN + tid * 32, 0, 0);

        #pragma unroll
        for (int c = 0; c < EPT; ++c) {
            int i = c * THREADS + tid;
            int n = i & (HDIM - 1);
            float h  = ((float*)s_h)[i];
            float sx = rsx[c];
            float su = rsu[c];
            bool fac = (n & 1) == 0;
            float a_std = fac ? 0.05f : (10.f / 1500.f);
            float a_stf = fac ? (10.f / 1500.f) : 0.05f;
            float u     = fac ? 0.15f : 0.45f;
            float sxn = sx + a_std * (1.f - sx) - 0.01f * su * sx * h;
            float sun = su + a_stf * (u - su)  + 0.01f * u * (1.f - su) * h;
            sxn = fminf(1.f, fmaxf(0.f, sxn));
            sun = fminf(1.f, fmaxf(0.f, sun));
            rsx[c] = sxn;
            rsu[c] = sun;
            ((_Float16*)s_hp)[i] = (_Float16)(sun * sxn * h);
        }
        for (int i = tid; i < BT * NIN; i += THREADS) {
            int m = i >> 7, k = i & (NIN - 1);
            ((_Float16*)s_x)[i] =
                (_Float16)g_x[((size_t)t * BATCH + b0 + m) * NIN + k];
        }
        __syncthreads();

        // ===== Phase G: pre = x@Win + hpost@Wrnn (WMMA), then h update =====
        {
            v8f acc[4] = {};
            v16h bq[2][4];    // double-buffered B fragments (parity = kt&1)

            // preload input-GEMM kt=0
            #pragma unroll
            for (int j = 0; j < 4; ++j)
                bq[0][j] = load_b_frag(wiT + (size_t)(nbase + j * 16 + l16) * NIN, 0, half);

            // input GEMM, K = 128 (4 K-tiles)
            #pragma unroll
            for (int kt = 0; kt < NIN / 32; ++kt) {
                if (kt < NIN / 32 - 1) {
                    #pragma unroll
                    for (int j = 0; j < 4; ++j)
                        bq[(kt + 1) & 1][j] = load_b_frag(
                            wiT + (size_t)(nbase + j * 16 + l16) * NIN, kt + 1, half);
                } else {
                    #pragma unroll
                    for (int j = 0; j < 4; ++j)
                        bq[(kt + 1) & 1][j] = load_b_frag(
                            wrT + (size_t)(nbase + j * 16 + l16) * HDIM, 0, half);
                }
                v16h a = load_a_frag(&s_x[l16][0], kt, half);
                #pragma unroll
                for (int j = 0; j < 4; ++j)
                    acc[j] = __builtin_amdgcn_wmma_f32_16x16x32_f16(
                        false, a, false, bq[kt & 1][j], (short)0, acc[j], false, false);
            }
            // recurrent GEMM, K = 512 (16 K-tiles); parity continues (4 is even)
            #pragma unroll
            for (int kt = 0; kt < HDIM / 32; ++kt) {
                if (kt < HDIM / 32 - 1) {
                    #pragma unroll
                    for (int j = 0; j < 4; ++j)
                        bq[(kt + 1) & 1][j] = load_b_frag(
                            wrT + (size_t)(nbase + j * 16 + l16) * HDIM, kt + 1, half);
                }
                v16h a = load_a_frag(&s_hp[l16][0], kt, half);
                #pragma unroll
                for (int j = 0; j < 4; ++j)
                    acc[j] = __builtin_amdgcn_wmma_f32_16x16x32_f16(
                        false, a, false, bq[kt & 1][j], (short)0, acc[j], false, false);
            }

            // h update directly in C/D layout: m = v + 8*half, n = nt*16 + l16
            #pragma unroll
            for (int j = 0; j < 4; ++j) {
                int n = nbase + j * 16 + l16;
                #pragma unroll
                for (int v = 0; v < 8; ++v) {
                    int m = v + 8 * half;
                    float pre  = acc[j][v] + bias[j];
                    float hold = s_h[m][n];
                    float nz   = g_noise[((size_t)t * BATCH + b0 + m) * HDIM + n];
                    float hn   = 0.9f * hold + 0.1f * pre + 0.05f * nz;
                    hn = hn > 0.f ? hn : 0.f;
                    s_h[m][n] = hn;
                    acc[j][v] = hn;            // keep h_new for the f16 spill
                }
            }
            __syncthreads();   // all hpost reads done -> safe to reuse s_hp
            #pragma unroll
            for (int j = 0; j < 4; ++j) {
                int n = nbase + j * 16 + l16;
                #pragma unroll
                for (int v = 0; v < 8; ++v)
                    s_hp[v + 8 * half][n] = (_Float16)acc[j][v];
            }
            __syncthreads();
        }

        // ============ Phase Y: y_t = h @ Wout (fused, waves 0..1) ============
        if (wave < 2) {
            v8f yacc = {};
            int n = wave * 16 + l16;
            v16h yb[2];
            yb[0] = load_b_frag(woT + (size_t)n * HDIM, 0, half);
            #pragma unroll
            for (int kt = 0; kt < HDIM / 32; ++kt) {
                if (kt < HDIM / 32 - 1)
                    yb[(kt + 1) & 1] = load_b_frag(woT + (size_t)n * HDIM, kt + 1, half);
                v16h a = load_a_frag(&s_hp[l16][0], kt, half);
                yacc = __builtin_amdgcn_wmma_f32_16x16x32_f16(
                    false, a, false, yb[kt & 1], (short)0, yacc, false, false);
            }
            #pragma unroll
            for (int v = 0; v < 8; ++v) {
                int m = v + 8 * half;
                g_y[((size_t)t * BATCH + b0 + m) * NOUT + n] = yacc[v] + bouty;
            }
        }
        __syncthreads();
    }
}

extern "C" void kernel_launch(void* const* d_in, const int* in_sizes, int n_in,
                              void* d_out, int out_size, void* d_ws, size_t ws_size,
                              hipStream_t stream) {
    const float* input_data = (const float*)d_in[0];
    const float* noise      = (const float*)d_in[1];
    const float* w_in       = (const float*)d_in[2];
    const float* w_rnn      = (const float*)d_in[3];
    const float* w_out      = (const float*)d_in[4];
    const float* b_rnn      = (const float*)d_in[5];
    const float* b_out      = (const float*)d_in[6];
    const float* h0         = (const float*)d_in[7];
    const float* sx0        = (const float*)d_in[8];
    const float* su0        = (const float*)d_in[9];

    _Float16* winT  = (_Float16*)d_ws;                 // [H][NIN]   128 KB
    _Float16* wrnnT = winT  + (size_t)HDIM * NIN;      // [H][H]     512 KB
    _Float16* woutT = wrnnT + (size_t)HDIM * HDIM;     // [NOUT][H]   32 KB

    int totalW = HDIM * NIN + HDIM * HDIM + NOUT * HDIM;
    prep_weights<<<(totalW + 255) / 256, 256, 0, stream>>>(
        w_in, w_rnn, w_out, winT, wrnnT, woutT);

    stp_rnn<<<NWG, THREADS, 0, stream>>>(
        input_data, noise, winT, wrnnT, woutT,
        b_rnn, b_out, h0, sx0, su0, (float*)d_out, /*zoff=*/0);
}
